// GnnJobActor_44452911514303
// MI455X (gfx1250) — matrix operand order repaired and checked by reading, not compiled, not run on hardware
//
#include <hip/hip_runtime.h>
#include <hip/hip_bf16.h>

#define BB 16
#define NN 2048
#define HH 128
#define NEGBIG -1e9f

typedef __attribute__((ext_vector_type(16))) __bf16 v16bf;
typedef __attribute__((ext_vector_type(8)))  float  v8f;
typedef __attribute__((ext_vector_type(4)))  unsigned int u32x4;
typedef __attribute__((ext_vector_type(4)))  int i32x4;

union FragU {
  v16bf v;
  u32x4 q[2];
  unsigned int w[8];
  unsigned short h[16];
};

static __device__ __forceinline__ unsigned short f2bf(float x) {
  union { __bf16 b; unsigned short u; } c;
  c.b = (__bf16)x;
  return c.u;
}
static __device__ __forceinline__ float bf2f(unsigned short u) {
  union { __bf16 b; unsigned short u; } c;
  c.u = u;
  return (float)c.b;
}

// -------- CDNA5 async global->LDS staging (guarded; ds_store fallback) ------
#if defined(__has_builtin)
#if __has_builtin(__builtin_amdgcn_global_load_async_to_lds_b128)
#define HAVE_ASYNC_LDS 1
#endif
#endif

static __device__ __forceinline__ void stage_seg(const unsigned short* gsrc,
                                                 unsigned short* ldst) {
#ifdef HAVE_ASYNC_LDS
  // prototype: (int4 addrspace(1)* src, int4 addrspace(3)* dst, imm off, imm cpol)
  __builtin_amdgcn_global_load_async_to_lds_b128(
      (__attribute__((address_space(1))) i32x4*)gsrc,
      (__attribute__((address_space(3))) i32x4*)ldst, 0, 0);
#else
  *(u32x4*)ldst = *(const u32x4*)gsrc;
#endif
}
static __device__ __forceinline__ void wait_async_2() {
#ifdef HAVE_ASYNC_LDS
#if __has_builtin(__builtin_amdgcn_s_wait_asynccnt)
  __builtin_amdgcn_s_wait_asynccnt(2);
#else
  asm volatile("s_wait_asynccnt 0x2" ::: "memory");
#endif
#endif
}
static __device__ __forceinline__ void wait_async_0() {
#ifdef HAVE_ASYNC_LDS
#if __has_builtin(__builtin_amdgcn_s_wait_asynccnt)
  __builtin_amdgcn_s_wait_asynccnt(0);
#else
  asm volatile("s_wait_asynccnt 0x0" ::: "memory");
#endif
#endif
}

// ---------------------------------------------------------------------------
// K1: bit-pack A^T (with self loops) + degree -> dis = rsqrt(deg).
// adjbits[(b*NN+i)*64 + w] bit t  <=>  A[b, j=w*32+t, i] != 0  (or j==i).
// Single HBM pass over the 268 MB int32 adjacency (the roofline floor).
// ---------------------------------------------------------------------------
__global__ __launch_bounds__(256) void k_pack(const int* __restrict__ adj,
                                              unsigned int* __restrict__ adjbits,
                                              float* __restrict__ dis) {
  int idx = blockIdx.x * blockDim.x + threadIdx.x;   // b*NN + i
  if (idx >= BB * NN) return;
  int b = idx >> 11;
  int i = idx & (NN - 1);
  const int* col = adj + (size_t)b * NN * NN + i;
  int deg = 0;
  for (int w = 0; w < 64; ++w) {
    unsigned int word = 0;
    int jbase = w << 5;
    #pragma unroll
    for (int t = 0; t < 32; ++t) {
      int j = jbase + t;
      int v = col[(size_t)j * NN];
      unsigned int bit = (v != 0 || j == i) ? 1u : 0u;
      word |= bit << t;
    }
    adjbits[(size_t)idx * 64 + w] = word;
    deg += __popc(word);
  }
  dis[idx] = rsqrtf((float)deg);
}

// ---------------------------------------------------------------------------
// K2: convert weights to bf16, transposed (Ho x K) for B-fragment loads.
// slot 0: W2^T, 1: W3^T, 2: A1[0:128,:]^T, 3: A2^T
// ---------------------------------------------------------------------------
__global__ __launch_bounds__(256) void k_prepw(const float* __restrict__ W2,
                                               const float* __restrict__ W3,
                                               const float* __restrict__ A1,
                                               const float* __restrict__ A2,
                                               unsigned short* __restrict__ wt) {
  int idx = blockIdx.x * blockDim.x + threadIdx.x;   // 4*128*128
  if (idx >= 4 * HH * HH) return;
  int mtx = idx >> 14;
  int r = (idx >> 7) & (HH - 1);  // output column h
  int k = idx & (HH - 1);         // reduction index
  float v;
  if (mtx == 0)      v = W2[k * HH + r];
  else if (mtx == 1) v = W3[k * HH + r];
  else if (mtx == 2) v = A1[k * HH + r];
  else               v = A2[k * HH + r];
  wt[mtx * HH * HH + r * HH + k] = f2bf(v);
}

// ---------------------------------------------------------------------------
// K3: layer-1 input transform: YT[b][h][j] = dis[b,j] * (f0*W1[0,h]+f1*W1[1,h])
// Stored transposed (H x N per batch) so propagation B-fragments are contiguous.
// ---------------------------------------------------------------------------
__global__ __launch_bounds__(256) void k_layer1(const float* __restrict__ feat,
                                                const float* __restrict__ W1,
                                                const float* __restrict__ dis,
                                                unsigned short* __restrict__ YT) {
  int idx = blockIdx.x * blockDim.x + threadIdx.x;   // B*N*H
  if (idx >= BB * NN * HH) return;
  int b = idx >> 18;
  int j = (idx >> 7) & (NN - 1);
  int h = idx & (HH - 1);
  size_t row = (size_t)b * NN + j;
  float f0 = feat[row * 2 + 0];
  float f1 = feat[row * 2 + 1];
  float v = (f0 * W1[h] + f1 * W1[HH + h]) * dis[row];
  YT[(size_t)b * (HH * NN) + (size_t)h * NN + j] = f2bf(v);
}

// ---------------------------------------------------------------------------
// K4: propagation  h_out = relu(dis[i] * (A^T @ Y) + bias), bf16 WMMA.
// 8 waves/block process 8 i-tiles of one batch; the 32x128 Y tile is shared
// through LDS (double-buffered, async global->LDS), cutting B-side L2 traffic
// 8x. A-fragments come from one adjacency bit-word per lane per iteration.
// ---------------------------------------------------------------------------
__global__ __launch_bounds__(256) void k_prop(const unsigned int* __restrict__ adjbits,
                                              const unsigned short* __restrict__ YT,
                                              const float* __restrict__ dis,
                                              const float* __restrict__ bias,
                                              unsigned short* __restrict__ out) {
  // 128 rows (h) x 32 j-values, rows padded to 40 halfwords (80B) so the 16
  // fragment lanes (stride 20 banks, gcd(20,64)=4 -> distinct) avoid conflicts.
  __shared__ unsigned short tile[2][HH * 40];

  int tid = threadIdx.x;
  int waveId = tid >> 5;
  int lane = tid & 31;
  int tileId = blockIdx.x * 8 + waveId;          // 0..2047
  int b = tileId >> 7;
  int i0 = (tileId & 127) << 4;
  int m = lane & 15;
  int hi = lane >> 4;       // lane half: selects K-subset / column half
  int kb = hi * 16;
  const unsigned int* abrow = adjbits + ((size_t)(b * NN + i0 + m)) * 64;
  const unsigned short* ytb = YT + (size_t)b * (HH * NN);

  // staging map: 512 x 16B segments per tile, 2 per thread
  int h0s = tid >> 2,         p0 = tid & 3;
  int h1s = (tid + 256) >> 2, p1 = tid & 3;      // +256 -> p unchanged, h += 64

  v8f zero = {0.f, 0.f, 0.f, 0.f, 0.f, 0.f, 0.f, 0.f};
  v8f acc[8];
  #pragma unroll
  for (int t = 0; t < 8; ++t) acc[t] = zero;

  // prologue: stage j-chunk 0 into buffer 0
  stage_seg(ytb + (size_t)h0s * NN + p0 * 8, &tile[0][h0s * 40 + p0 * 8]);
  stage_seg(ytb + (size_t)h1s * NN + p1 * 8, &tile[0][h1s * 40 + p1 * 8]);

  unsigned int word = abrow[0];
  for (int it = 0; it < 64; ++it) {
    int cur = it & 1;
    if (it < 63) {
      int jb = (it + 1) << 5;
      stage_seg(ytb + (size_t)h0s * NN + jb + p0 * 8, &tile[cur ^ 1][h0s * 40 + p0 * 8]);
      stage_seg(ytb + (size_t)h1s * NN + jb + p1 * 8, &tile[cur ^ 1][h1s * 40 + p1 * 8]);
      wait_async_2();           // current buffer's 2 loads done (in-order)
    } else {
      wait_async_0();
    }
    __syncthreads();

    unsigned int wnext = (it < 63) ? abrow[it + 1] : 0u;   // prefetch next word

    FragU a;
    #pragma unroll
    for (int e = 0; e < 16; ++e) {
      int bit = e + kb;   // A 16x32 bf16 fragment K layout (contiguous k = 16*hi + e)
      a.h[e] = ((word >> bit) & 1u) ? (unsigned short)0x3F80 : (unsigned short)0;
    }

    FragU fb[8];
    #pragma unroll
    for (int t = 0; t < 8; ++t) {
      const u32x4* p = reinterpret_cast<const u32x4*>(&tile[cur][(t * 16 + m) * 40 + hi * 16]);
      fb[t].q[0] = p[0];
      fb[t].q[1] = p[1];
    }
    #pragma unroll
    for (int t = 0; t < 8; ++t)
      acc[t] = __builtin_amdgcn_wmma_f32_16x16x32_bf16(
          false, a.v, false, fb[t].v, (short)0, acc[t], false, false);

    word = wnext;
    __syncthreads();
  }

  #pragma unroll
  for (int r = 0; r < 8; ++r) {
    int mm = r + hi * 8;
    int row = b * NN + i0 + mm;
    float d = dis[row];
    #pragma unroll
    for (int t = 0; t < 8; ++t) {
      int h = t * 16 + m;
      float v = fmaxf(d * acc[t][r] + bias[h], 0.0f);
      out[(size_t)row * HH + h] = f2bf(v);
    }
  }
}

// ---------------------------------------------------------------------------
// K5: generic 128x128 bf16 WMMA GEMM over all B*N rows.
//   out = act( X @ W  (+ bias) ) * scale, row-major or transposed-per-batch.
// B fragments are preloaded per K-step so the 8 WMMAs issue back-to-back.
// ---------------------------------------------------------------------------
__global__ __launch_bounds__(256) void k_gemm(const unsigned short* __restrict__ X,
                                              const unsigned short* __restrict__ WT,
                                              unsigned short* __restrict__ out,
                                              const float* __restrict__ scale,
                                              const float* __restrict__ bias,
                                              int biasPerBatch, int relu, int transposeOut) {
  int wave = (blockIdx.x * blockDim.x + threadIdx.x) >> 5;  // 0..2047
  int lane = threadIdx.x & 31;
  int row0 = wave << 4;
  int b = row0 >> 11;
  int m = lane & 15;
  int hi = lane >> 4;
  const unsigned short* xrow = X + (size_t)(row0 + m) * HH;

  v8f zero = {0.f, 0.f, 0.f, 0.f, 0.f, 0.f, 0.f, 0.f};
  v8f acc[8];
  #pragma unroll
  for (int t = 0; t < 8; ++t) acc[t] = zero;

  #pragma unroll
  for (int it = 0; it < 4; ++it) {
    int k0 = it << 5;
    FragU a;
    const u32x4* pa = reinterpret_cast<const u32x4*>(xrow + k0 + hi * 16);
    a.q[0] = pa[0];      // K = kb .. kb+7
    a.q[1] = pa[1];      // K = kb+8 .. kb+15
    int kbB = k0 + hi * 16;
    FragU fb[8];
    #pragma unroll
    for (int t = 0; t < 8; ++t) {
      const u32x4* pb = reinterpret_cast<const u32x4*>(WT + (size_t)(t * 16 + m) * HH + kbB);
      fb[t].q[0] = pb[0];
      fb[t].q[1] = pb[1];
    }
    #pragma unroll
    for (int t = 0; t < 8; ++t)
      acc[t] = __builtin_amdgcn_wmma_f32_16x16x32_bf16(
          false, a.v, false, fb[t].v, (short)0, acc[t], false, false);
  }

  #pragma unroll
  for (int r = 0; r < 8; ++r) {
    int mm = r + hi * 8;
    int row = row0 + mm;
    float sc = scale ? scale[row] : 1.0f;
    #pragma unroll
    for (int t = 0; t < 8; ++t) {
      int h = t * 16 + m;
      float v = acc[t][r];
      if (bias) v += biasPerBatch ? bias[b * HH + h] : bias[h];
      v *= sc;
      if (relu) v = fmaxf(v, 0.0f);
      if (transposeOut)
        out[(size_t)b * (HH * NN) + (size_t)h * NN + (row & (NN - 1))] = f2bf(v);
      else
        out[(size_t)row * HH + h] = f2bf(v);
    }
  }
}

// ---------------------------------------------------------------------------
// K6: global mean pool over N
// ---------------------------------------------------------------------------
__global__ __launch_bounds__(128) void k_pool(const unsigned short* __restrict__ X,
                                              float* __restrict__ pool) {
  int b = blockIdx.x, h = threadIdx.x;
  float s = 0.f;
  for (int i = 0; i < NN; ++i) s += bf2f(X[((size_t)(b * NN + i)) * HH + h]);
  pool[b * HH + h] = s * (1.0f / NN);
}

// ---------------------------------------------------------------------------
// K7: per-batch actor bias: ab1 + pool@A1[128:256] + ms@A1[256:384]
// ---------------------------------------------------------------------------
__global__ __launch_bounds__(128) void k_bias2(const float* __restrict__ pool,
                                               const float* __restrict__ A1,
                                               const float* __restrict__ ab1,
                                               const float* __restrict__ ms,
                                               float* __restrict__ bias2) {
  int b = blockIdx.x, h = threadIdx.x;
  float s = ab1[h];
  for (int k = 0; k < HH; ++k) s += pool[b * HH + k] * A1[(HH + k) * HH + h];
  for (int k = 0; k < HH; ++k) s += ms[k] * A1[(2 * HH + k) * HH + h];
  bias2[b * HH + h] = s;
}

// ---------------------------------------------------------------------------
// K8: scores = (a2 @ A3 + ab3) * 10
// ---------------------------------------------------------------------------
__global__ __launch_bounds__(256) void k_scores(const unsigned short* __restrict__ a2,
                                                const float* __restrict__ A3,
                                                const float* __restrict__ ab3,
                                                float* __restrict__ scores) {
  int row = blockIdx.x * blockDim.x + threadIdx.x;
  if (row >= BB * NN) return;
  float s = ab3[0];
  for (int k = 0; k < HH; ++k) s += bf2f(a2[(size_t)row * HH + k]) * A3[k];
  scores[row] = s * 10.0f;
}

// ---------------------------------------------------------------------------
// K9: per-batch softmax -> masked log-softmax -> logp/entropy + critic value
// ---------------------------------------------------------------------------
static __device__ float blk_max(float v, float* red) {
  int tid = threadIdx.x;
  red[tid] = v; __syncthreads();
  for (int s = 128; s > 0; s >>= 1) {
    if (tid < s) red[tid] = fmaxf(red[tid], red[tid + s]);
    __syncthreads();
  }
  float r = red[0]; __syncthreads();
  return r;
}
static __device__ float blk_sum(float v, float* red) {
  int tid = threadIdx.x;
  red[tid] = v; __syncthreads();
  for (int s = 128; s > 0; s >>= 1) {
    if (tid < s) red[tid] = red[tid] + red[tid + s];
    __syncthreads();
  }
  float r = red[0]; __syncthreads();
  return r;
}

__global__ __launch_bounds__(256) void k_final(const float* __restrict__ scores,
                                               const int* __restrict__ cand,
                                               const int* __restrict__ action,
                                               const float* __restrict__ pool,
                                               const float* __restrict__ C1,
                                               const float* __restrict__ cb1,
                                               const float* __restrict__ C2,
                                               const float* __restrict__ cb2,
                                               float* __restrict__ out) {
  __shared__ float red[256];
  int b = blockIdx.x, tid = threadIdx.x;
  float s[8];
  #pragma unroll
  for (int u = 0; u < 8; ++u) s[u] = scores[b * NN + u * 256 + tid];

  float lmax = -1e30f;
  #pragma unroll
  for (int u = 0; u < 8; ++u) lmax = fmaxf(lmax, s[u]);
  float m1 = blk_max(lmax, red);

  float lsum = 0.f;
  #pragma unroll
  for (int u = 0; u < 8; ++u) lsum += expf(s[u] - m1);
  float Z = blk_sum(lsum, red);

  float l[8];
  #pragma unroll
  for (int u = 0; u < 8; ++u) {
    int i = u * 256 + tid;
    float p = expf(s[u] - m1) / Z;
    l[u] = (cand[b * NN + i] != 0) ? p : NEGBIG;
  }

  float lmax2 = NEGBIG;
  #pragma unroll
  for (int u = 0; u < 8; ++u) lmax2 = fmaxf(lmax2, l[u]);
  float m2 = blk_max(lmax2, red);

  float s2 = 0.f;
  #pragma unroll
  for (int u = 0; u < 8; ++u) s2 += expf(l[u] - m2);
  float sum2 = blk_sum(s2, red);
  float lse = logf(sum2);

  int act = action[b];
  float ent = 0.f;
  #pragma unroll
  for (int u = 0; u < 8; ++u) {
    int i = u * 256 + tid;
    float lp = l[u] - m2 - lse;
    float p = expf(lp);
    ent -= p * lp;
    if (i == act) out[BB + b] = lp;          // log_prob
  }
  float entt = blk_sum(ent, red);

  // critic: value = relu(pool @ C1 + cb1) @ C2 + cb2
  float cv = 0.f;
  if (tid < 32) {
    float a = cb1[tid];
    for (int k = 0; k < HH; ++k) a += pool[b * HH + k] * C1[k * 32 + tid];
    cv = fmaxf(a, 0.f) * C2[tid];
  }
  float val = blk_sum(cv, red);

  if (tid == 0) {
    out[b] = (float)act;                     // action
    out[2 * BB + b] = entt;                  // entropy
    out[3 * BB + b] = val + cb2[0];          // value
  }
}

// ---------------------------------------------------------------------------
extern "C" void kernel_launch(void* const* d_in, const int* in_sizes, int n_in,
                              void* d_out, int out_size, void* d_ws, size_t ws_size,
                              hipStream_t stream) {
  (void)in_sizes; (void)n_in; (void)out_size; (void)ws_size;
  const float* features = (const float*)d_in[0];
  const int*   adj      = (const int*)d_in[1];
  const int*   cand     = (const int*)d_in[2];
  const int*   action   = (const int*)d_in[3];
  const float* W1  = (const float*)d_in[4];
  const float* b1  = (const float*)d_in[5];
  const float* W2  = (const float*)d_in[6];
  const float* b2  = (const float*)d_in[7];
  const float* W3  = (const float*)d_in[8];
  const float* b3  = (const float*)d_in[9];
  const float* ms  = (const float*)d_in[10];
  const float* A1  = (const float*)d_in[11];
  const float* ab1 = (const float*)d_in[12];
  const float* A2  = (const float*)d_in[13];
  const float* ab2 = (const float*)d_in[14];
  const float* A3  = (const float*)d_in[15];
  const float* ab3 = (const float*)d_in[16];
  const float* C1  = (const float*)d_in[17];
  const float* cb1 = (const float*)d_in[18];
  const float* C2  = (const float*)d_in[19];
  const float* cb2 = (const float*)d_in[20];
  float* out = (float*)d_out;

  char* ws = (char*)d_ws;
  constexpr size_t SZ_BITS = (size_t)BB * NN * 64 * 4;        // 8 MB (reused as a2)
  constexpr size_t SZ_BF   = (size_t)BB * NN * HH * 2;        // 8 MB
  unsigned int*   adjbits = (unsigned int*)(ws);
  unsigned short* Xbuf    = (unsigned short*)(ws + SZ_BITS);            // h (row-major)
  unsigned short* YTbuf   = (unsigned short*)(ws + SZ_BITS + SZ_BF);    // Y^T / a1
  char* tail = ws + SZ_BITS + 2 * SZ_BF;
  float*          dis     = (float*)(tail);                    tail += (size_t)BB * NN * 4;
  float*          scores  = (float*)(tail);                    tail += (size_t)BB * NN * 4;
  unsigned short* wt      = (unsigned short*)(tail);           tail += 4 * HH * HH * 2;
  float*          pool    = (float*)(tail);                    tail += BB * HH * 4;
  float*          bias2   = (float*)(tail);
  unsigned short* a2buf   = (unsigned short*)adjbits;          // reuse after last prop

  const unsigned short* W2T  = wt;
  const unsigned short* W3T  = wt + 1 * HH * HH;
  const unsigned short* A1aT = wt + 2 * HH * HH;
  const unsigned short* A2T  = wt + 3 * HH * HH;

  // 1. pack adjacency bits + dis (single HBM pass over 268 MB)
  k_pack<<<(BB * NN) / 256, 256, 0, stream>>>(adj, adjbits, dis);
  // 2. weights -> bf16 transposed
  k_prepw<<<(4 * HH * HH) / 256, 256, 0, stream>>>(W2, W3, A1, A2, wt);
  // 3. layer 1: Y1^T = dis * (features @ W1)
  k_layer1<<<(BB * NN * HH) / 256, 256, 0, stream>>>(features, W1, dis, YTbuf);
  // 4. layer 1 propagation
  k_prop<<<256, 256, 0, stream>>>(adjbits, YTbuf, dis, b1, Xbuf);
  // 5. layer 2: Y2^T = dis * (h @ W2) ; propagate
  k_gemm<<<256, 256, 0, stream>>>(Xbuf, W2T, YTbuf, dis, nullptr, 0, 0, 1);
  k_prop<<<256, 256, 0, stream>>>(adjbits, YTbuf, dis, b2, Xbuf);
  // 6. layer 3
  k_gemm<<<256, 256, 0, stream>>>(Xbuf, W3T, YTbuf, dis, nullptr, 0, 0, 1);
  k_prop<<<256, 256, 0, stream>>>(adjbits, YTbuf, dis, b3, Xbuf);
  // 7. pooling + fused actor bias (pool/machine_state parts of A1)
  k_pool<<<BB, HH, 0, stream>>>(Xbuf, pool);
  k_bias2<<<BB, HH, 0, stream>>>(pool, A1, ab1, ms, bias2);
  // 8. actor MLP (WMMA): a1 = relu(h@A1a + bias2[b]); a2 = relu(a1@A2 + ab2)
  k_gemm<<<256, 256, 0, stream>>>(Xbuf, A1aT, YTbuf, nullptr, bias2, 1, 1, 0);
  k_gemm<<<256, 256, 0, stream>>>(YTbuf, A2T, a2buf, nullptr, ab2, 0, 1, 0);
  // 9. scores + policy head + critic
  k_scores<<<(BB * NN) / 256, 256, 0, stream>>>(a2buf, A3, ab3, scores);
  k_final<<<BB, 256, 0, stream>>>(scores, cand, action, pool, C1, cb1, C2, cb2, out);
}
